// QWenAttention_51350628990981
// MI455X (gfx1250) — matrix operand order repaired and checked by reading, not compiled, not run on hardware
//
#include <hip/hip_runtime.h>
#include <hip/hip_bf16.h>

// ---------------------------------------------------------------------------
// QWen-style attention layer for MI455X (gfx1250, wave32).
// All matmuls via v_wmma_f32_16x16x32_bf16; all LDS staging via
// global_load_async_to_lds_b128 (ASYNCcnt) with double-buffered GEMM tiles.
// ---------------------------------------------------------------------------

typedef __attribute__((ext_vector_type(16))) __bf16 v16bf;
typedef __attribute__((ext_vector_type(8)))  float  v8f;

#define HIDDEN    4096
#define NUM_HEADS 32
#define HEAD_DIM  128
#define SEQ       2048
#define BATCH     2
#define NTOK      (BATCH * SEQ)          // 4096
#define QKV_N     (3 * HIDDEN)           // 12288
#define ATT_SCALE 0.08838834764831845f   // 128^-0.5

// ---------------------------------------------------------------------------
// CDNA5 async memory->LDS copy (ASYNCcnt) helpers
// ---------------------------------------------------------------------------
__device__ inline unsigned lds_addr32(const void* p) {
  // Flat LDS aperture: low 32 bits of a generic pointer to __shared__ are the
  // LDS byte address (ISA 10.2 per-aperture calc).
  return (unsigned)(unsigned long long)p;
}

__device__ inline void async_copy_b128(unsigned lds_dst, const void* gsrc) {
  asm volatile("global_load_async_to_lds_b128 %0, %1, off"
               :
               : "v"(lds_dst), "v"(gsrc)
               : "memory");
}

__device__ inline void wait_async0() {
  asm volatile("s_wait_asynccnt 0x0" ::: "memory");
}

// ---------------------------------------------------------------------------
// WMMA helpers
// ---------------------------------------------------------------------------
__device__ inline v8f wmma_bf16(v16bf a, v16bf b, v8f c) {
  return __builtin_amdgcn_wmma_f32_16x16x32_bf16(false, a, false, b, (short)0, c,
                                                 false, false);
}

__device__ inline v8f zero8() {
  v8f z;
  #pragma unroll
  for (int i = 0; i < 8; ++i) z[i] = 0.0f;
  return z;
}

// A-fragment (16x32 bf16, M x K). Lane m = L%16, half = L/16.
// VGPR v holds pair; k = (v>=4 ? 16:0) + half*8 + (v%4)*2 + p.
// tile row-major [m][k], row stride `ld` elements.
__device__ inline v16bf frag_a(const __bf16* tile, int ld, int lane) {
  const int m    = lane & 15;
  const int half = (lane >> 4) << 3;
  const __bf16* row = tile + m * ld + half;
  union { unsigned u[8]; v16bf v; } cvt;
  #pragma unroll
  for (int g = 0; g < 2; ++g)
    #pragma unroll
    for (int j = 0; j < 4; ++j)
      cvt.u[g * 4 + j] = *(const unsigned*)(row + g * 16 + j * 2);
  return cvt.v;
}

// B-fragment (32x16 bf16, K x N). Lane n = L%16, kh = L/16; e -> k = (e>=8 ? 16:0) + kh*8 + e%8.
// tileT stored TRANSPOSED [n][k], row stride `ld`; two 8-bf16 chunks/lane.
__device__ inline v16bf frag_b(const __bf16* tileT, int ld, int lane) {
  const int n  = lane & 15;
  const int kh = (lane >> 4) << 3;
  const uint4* p = (const uint4*)(tileT + n * ld + kh);
  union { uint4 u[2]; v16bf v; } cvt;
  cvt.u[0] = p[0];
  cvt.u[1] = p[2];
  return cvt.v;
}

// ---------------------------------------------------------------------------
// f32 -> bf16 elementwise convert (for X)
// ---------------------------------------------------------------------------
__global__ void cvt_bf16_kernel(const float* __restrict__ in,
                                __bf16* __restrict__ out, size_t n) {
  size_t i = (size_t)blockIdx.x * blockDim.x + threadIdx.x;
  size_t stride = (size_t)gridDim.x * blockDim.x;
  for (; i < n; i += stride) out[i] = (__bf16)in[i];
}

// ---------------------------------------------------------------------------
// f32 [K][N] -> bf16 [N][K] tiled transpose-convert (for weights)
// Block 256 = 32x8, tile 32x32.
// ---------------------------------------------------------------------------
__global__ void __launch_bounds__(256)
transpose_bf16_kernel(const float* __restrict__ in, __bf16* __restrict__ out,
                      int K, int N) {
  __shared__ float tile[32][33];
  const int k0 = blockIdx.y * 32, n0 = blockIdx.x * 32;
  const int tx = threadIdx.x & 31, ty = threadIdx.x >> 5;
  #pragma unroll
  for (int r = 0; r < 4; ++r)
    tile[ty + r * 8][tx] = in[(size_t)(k0 + ty + r * 8) * N + n0 + tx];
  __syncthreads();
  #pragma unroll
  for (int r = 0; r < 4; ++r)
    out[(size_t)(n0 + ty + r * 8) * K + k0 + tx] = (__bf16)tile[tx][ty + r * 8];
}

// ---------------------------------------------------------------------------
// Tiled bf16 WMMA GEMM:  C[M,N] (f32) = A[M,K] @ Wt[N,K]^T (+ bias[N])
// A row-major, weights PRE-TRANSPOSED [N][K] so both LDS panels stage with
// contiguous async b128 copies.  Block 256 threads = 8 waves (4Mx2N),
// block tile 128x128, wave tile 32x64, K-step 32, double-buffered LDS.
// ---------------------------------------------------------------------------
__global__ void __launch_bounds__(256)
gemm_bf16_kernel(const __bf16* __restrict__ A, const __bf16* __restrict__ Wt,
                 const float* __restrict__ bias, float* __restrict__ C,
                 int M, int N, int K) {
  __shared__ __align__(16) __bf16 As[2][128 * 32];  // [m][k]
  __shared__ __align__(16) __bf16 Bt[2][128 * 32];  // [n][k] (pre-transposed)

  const int tid  = threadIdx.x;
  const int lane = tid & 31;
  const int wave = tid >> 5;
  const int mw   = wave >> 1;                       // 0..3
  const int nw   = wave & 1;                        // 0..1
  const int mBase = blockIdx.y * 128;
  const int nBase = blockIdx.x * 128;

  v8f acc[2][4];
  #pragma unroll
  for (int j = 0; j < 4; ++j) {
    float bv = bias ? bias[nBase + nw * 64 + j * 16 + (lane & 15)] : 0.0f;
    #pragma unroll
    for (int r = 0; r < 8; ++r) { acc[0][j][r] = bv; acc[1][j][r] = bv; }
  }

  const int sRow = tid >> 1;                        // 0..127 (row / n index)
  const int sOff = (tid & 1) * 16;                  // 0 or 16 elements

  auto stage = [&](int buf, int k0) {
    const __bf16* ga = A  + (size_t)(mBase + sRow) * K + k0 + sOff;
    const __bf16* gb = Wt + (size_t)(nBase + sRow) * K + k0 + sOff;
    unsigned la = lds_addr32(&As[buf][sRow * 32 + sOff]);
    unsigned lb = lds_addr32(&Bt[buf][sRow * 32 + sOff]);
    async_copy_b128(la, ga);
    async_copy_b128(la + 16, ga + 8);
    async_copy_b128(lb, gb);
    async_copy_b128(lb + 16, gb + 8);
  };

  const int nIter = K >> 5;
  stage(0, 0);
  for (int it = 0; it < nIter; ++it) {
    wait_async0();            // own async copies done
    __syncthreads();          // everyone's copies done -> tile ready
    if (it + 1 < nIter) stage((it + 1) & 1, (it + 1) << 5);  // overlap

    const int buf = it & 1;
    v16bf a0 = frag_a(&As[buf][(mw * 32) * 32], 32, lane);
    v16bf a1 = frag_a(&As[buf][(mw * 32 + 16) * 32], 32, lane);
    #pragma unroll
    for (int j = 0; j < 4; ++j) {
      v16bf b = frag_b(&Bt[buf][(nw * 64 + j * 16) * 32], 32, lane);
      acc[0][j] = wmma_bf16(a0, b, acc[0][j]);
      acc[1][j] = wmma_bf16(a1, b, acc[1][j]);
    }
    __syncthreads();          // reads done before buf is overwritten (it+2)
  }

  // C/D layout: VGPR r -> row m = r + (lane/16)*8 ; col n = lane%16
  const int mOff = (lane >> 4) << 3;
  #pragma unroll
  for (int i = 0; i < 2; ++i)
    #pragma unroll
    for (int j = 0; j < 4; ++j)
      #pragma unroll
      for (int r = 0; r < 8; ++r) {
        int m = mBase + mw * 32 + i * 16 + r + mOff;
        int n = nBase + nw * 64 + j * 16 + (lane & 15);
        C[(size_t)m * N + n] = acc[i][j][r];
      }
}

// ---------------------------------------------------------------------------
// RoPE (NeoX) + bf16 pack.  qkv: [NTOK, 12288] f32.
// Qr/Kr: [NTOK, HIDDEN] bf16 (col = h*128+d).  Vt_g: [B,H,D,S] bf16
// (transposed so attention V tiles stage with contiguous async copies).
// ---------------------------------------------------------------------------
__global__ void __launch_bounds__(64)
rope_kernel(const float* __restrict__ qkv, const int* __restrict__ positions,
            __bf16* __restrict__ Qr, __bf16* __restrict__ Kr,
            __bf16* __restrict__ Vt_g) {
  const int tok = blockIdx.x;   // 0..NTOK-1
  const int h   = blockIdx.y;   // 0..31
  const int d   = threadIdx.x;  // 0..63 (half of HEAD_DIM)
  const int b   = tok / SEQ;
  const int s   = tok - b * SEQ;
  const float pos = (float)positions[tok];
  const float inv = __powf(10000.0f, -(float)d / 64.0f);
  float sv, cv;
  __sincosf(pos * inv, &sv, &cv);

  const float* q = qkv + (size_t)tok * QKV_N + h * HEAD_DIM;
  const float* k = q + HIDDEN;
  const float* v = q + 2 * HIDDEN;
  const size_t ob = (size_t)tok * HIDDEN + h * HEAD_DIM;

  float q1 = q[d], q2 = q[d + 64];
  Qr[ob + d]      = (__bf16)(q1 * cv - q2 * sv);
  Qr[ob + d + 64] = (__bf16)(q2 * cv + q1 * sv);
  float k1 = k[d], k2 = k[d + 64];
  Kr[ob + d]      = (__bf16)(k1 * cv - k2 * sv);
  Kr[ob + d + 64] = (__bf16)(k2 * cv + k1 * sv);

  const size_t vb = ((size_t)(b * NUM_HEADS + h) * HEAD_DIM) * SEQ + s;
  Vt_g[vb + (size_t)d * SEQ]        = (__bf16)v[d];
  Vt_g[vb + (size_t)(d + 64) * SEQ] = (__bf16)v[d + 64];
}

// ---------------------------------------------------------------------------
// Causal flash attention, all-WMMA, async LDS staging.
// Block = 128 threads (4 waves) = one (b, h, 64-row q-block).
// ---------------------------------------------------------------------------
__global__ void __launch_bounds__(128)
attn_kernel(const __bf16* __restrict__ Qr, const __bf16* __restrict__ Kr,
            const __bf16* __restrict__ Vt_g, __bf16* __restrict__ Out) {
  const int qBase = blockIdx.x * 64;
  const int h     = blockIdx.y;
  const int b     = blockIdx.z;
  const int tid = threadIdx.x, lane = tid & 31, wave = tid >> 5;

  __shared__ __align__(16) __bf16 Qs[64 * 128];   // [qrow][d]
  __shared__ __align__(16) __bf16 Ks[32 * 128];   // [key][d]  (B for scores)
  __shared__ __align__(16) __bf16 Vt[128 * 32];   // [d][key]  (B for PV)
  __shared__ float  Sc[4][16 * 32];
  __shared__ __align__(4) __bf16 Ps[4][16 * 32];
  __shared__ float  mRun[4][16], lRun[4][16], rScale[4][16];

  const size_t bhBase = (size_t)b * SEQ * HIDDEN + h * HEAD_DIM;
  const size_t vBase  = (size_t)(b * NUM_HEADS + h) * HEAD_DIM * SEQ;

  // --- stage Q block (64x128) via async copies: 8 x b128 per thread ---
  {
    const int row = tid >> 1, half = (tid & 1) * 64;
    const __bf16* g = Qr + bhBase + (size_t)(qBase + row) * HIDDEN + half;
    unsigned l = lds_addr32(&Qs[row * 128 + half]);
    #pragma unroll
    for (int j = 0; j < 8; ++j) async_copy_b128(l + j * 16, g + j * 8);
  }
  if (lane < 16) {
    mRun[wave][lane] = -1e30f;
    lRun[wave][lane] = 0.0f;
  }
  wait_async0();
  __syncthreads();

  v16bf qf[4];
  #pragma unroll
  for (int dc = 0; dc < 4; ++dc)
    qf[dc] = frag_a(&Qs[wave * 16 * 128] + dc * 32, 128, lane);

  v8f acc[8];
  #pragma unroll
  for (int j = 0; j < 8; ++j) acc[j] = zero8();

  const int kvEnd = qBase + 64;                   // causal upper bound
  const int qwMax = qBase + wave * 16 + 15;
  const int mOff  = (lane >> 4) << 3;

  const int kRow = tid >> 2, kQt = (tid & 3) * 32;  // K staging coords

  for (int kv = 0; kv < kvEnd; kv += 32) {
    // --- stage K tile [32 keys][128 d]: 4 x b128 per thread ---
    {
      const __bf16* g = Kr + bhBase + (size_t)(kv + kRow) * HIDDEN + kQt;
      unsigned l = lds_addr32(&Ks[kRow * 128 + kQt]);
      #pragma unroll
      for (int j = 0; j < 4; ++j) async_copy_b128(l + j * 16, g + j * 8);
    }
    // --- stage V tile [128 d][32 keys] from transposed global V ---
    {
      const __bf16* g = Vt_g + vBase + (size_t)tid * SEQ + kv;
      unsigned l = lds_addr32(&Vt[tid * 32]);
      #pragma unroll
      for (int j = 0; j < 4; ++j) async_copy_b128(l + j * 16, g + j * 8);
    }
    wait_async0();
    __syncthreads();

    if (kv <= qwMax) {                            // wave has live keys here
      float* sc = Sc[wave];
      // --- scores: two 16x16 tiles, 4 WMMAs each over D=128 ---
      #pragma unroll
      for (int kh2 = 0; kh2 < 2; ++kh2) {
        v8f s = zero8();
        #pragma unroll
        for (int dc = 0; dc < 4; ++dc) {
          v16bf bk = frag_b(&Ks[(kh2 * 16) * 128 + dc * 32], 128, lane);
          s = wmma_bf16(qf[dc], bk, s);
        }
        const int qRow0 = qBase + wave * 16 + mOff;
        const int key   = kv + kh2 * 16 + (lane & 15);
        #pragma unroll
        for (int r = 0; r < 8; ++r) {
          float val = s[r] * ATT_SCALE;
          if (key > qRow0 + r) val = -1e30f;      // causal mask (finite)
          sc[(r + mOff) * 32 + kh2 * 16 + (lane & 15)] = val;
        }
      }
      // --- online softmax: lanes 0..15 each own one q-row ---
      if (lane < 16) {
        const int row = lane;
        float mOld = mRun[wave][row];
        float mNew = mOld;
        #pragma unroll
        for (int jj = 0; jj < 32; ++jj)
          mNew = fmaxf(mNew, sc[row * 32 + jj]);
        float corr = __expf(mOld - mNew);
        float sum = 0.0f;
        #pragma unroll
        for (int jj = 0; jj < 32; ++jj) {
          float p = __expf(sc[row * 32 + jj] - mNew);
          sum += p;
          Ps[wave][row * 32 + jj] = (__bf16)p;
        }
        mRun[wave][row]   = mNew;
        lRun[wave][row]   = lRun[wave][row] * corr + sum;
        rScale[wave][row] = corr;
      }
      // --- rescale accumulators ---
      #pragma unroll
      for (int r = 0; r < 8; ++r) {
        float corr = rScale[wave][r + mOff];
        #pragma unroll
        for (int j = 0; j < 8; ++j) acc[j][r] *= corr;
      }
      // --- P @ V : 8 WMMAs cover D=128 ---
      v16bf pf = frag_a(Ps[wave], 32, lane);
      #pragma unroll
      for (int j = 0; j < 8; ++j) {
        v16bf bv = frag_b(&Vt[(j * 16) * 32], 32, lane);
        acc[j] = wmma_bf16(pf, bv, acc[j]);
      }
    }
    __syncthreads();
  }

  // --- finalize ---
  float invL[8];
  #pragma unroll
  for (int r = 0; r < 8; ++r) invL[r] = 1.0f / lRun[wave][r + mOff];
  #pragma unroll
  for (int j = 0; j < 8; ++j)
    #pragma unroll
    for (int r = 0; r < 8; ++r) {
      int m = qBase + wave * 16 + r + mOff;
      int d = j * 16 + (lane & 15);
      Out[bhBase + (size_t)m * HIDDEN + d] = (__bf16)(acc[j][r] * invL[r]);
    }
}

// ---------------------------------------------------------------------------
// Host-side launcher.  Workspace (bytes):
//   Xb 32MB | Wqt 96MB | Wpt 32MB | QKV 192MB | Qr 32MB | Kr 32MB
//   | Vt_g 32MB | Attn 32MB    (~480MB)
// ---------------------------------------------------------------------------
extern "C" void kernel_launch(void* const* d_in, const int* in_sizes, int n_in,
                              void* d_out, int out_size, void* d_ws,
                              size_t ws_size, hipStream_t stream) {
  (void)in_sizes; (void)n_in; (void)out_size; (void)ws_size;

  const int*   positions = (const int*)d_in[0];
  const float* hidden    = (const float*)d_in[1];
  const float* w_qkv     = (const float*)d_in[2];
  const float* b_qkv     = (const float*)d_in[3];
  const float* w_proj    = (const float*)d_in[4];
  float*       out       = (float*)d_out;

  const size_t nX  = (size_t)NTOK * HIDDEN;
  const size_t nWq = (size_t)HIDDEN * QKV_N;
  const size_t nWp = (size_t)HIDDEN * HIDDEN;

  char* ws = (char*)d_ws;
  __bf16* Xb   = (__bf16*)ws;  ws += nX  * 2;
  __bf16* Wqt  = (__bf16*)ws;  ws += nWq * 2;   // [12288][4096] bf16
  __bf16* Wpt  = (__bf16*)ws;  ws += nWp * 2;   // [4096][4096] bf16
  float*  QKV  = (float*)ws;   ws += nWq * 4;   // [NTOK][12288] f32
  __bf16* Qr   = (__bf16*)ws;  ws += nX * 2;
  __bf16* Kr   = (__bf16*)ws;  ws += nX * 2;
  __bf16* Vt_g = (__bf16*)ws;  ws += nX * 2;    // [B][H][D][S]
  __bf16* Attn = (__bf16*)ws;  ws += nX * 2;

  // 1) convert X; transpose-convert weights to [N][K] bf16
  cvt_bf16_kernel<<<2048, 256, 0, stream>>>(hidden, Xb, nX);
  transpose_bf16_kernel<<<dim3(QKV_N / 32, HIDDEN / 32), 256, 0, stream>>>(
      w_qkv, Wqt, HIDDEN, QKV_N);
  transpose_bf16_kernel<<<dim3(HIDDEN / 32, HIDDEN / 32), 256, 0, stream>>>(
      w_proj, Wpt, HIDDEN, HIDDEN);

  // 2) QKV projection
  gemm_bf16_kernel<<<dim3(QKV_N / 128, NTOK / 128), 256, 0, stream>>>(
      Xb, Wqt, b_qkv, QKV, NTOK, QKV_N, HIDDEN);

  // 3) RoPE + bf16 pack (V written globally transposed)
  rope_kernel<<<dim3(NTOK, NUM_HEADS), 64, 0, stream>>>(QKV, positions, Qr, Kr,
                                                        Vt_g);

  // 4) causal flash attention
  attn_kernel<<<dim3(SEQ / 64, NUM_HEADS, BATCH), 128, 0, stream>>>(Qr, Kr,
                                                                    Vt_g, Attn);

  // 5) output projection -> d_out (f32)
  gemm_bf16_kernel<<<dim3(HIDDEN / 128, NTOK / 128), 256, 0, stream>>>(
      Attn, Wpt, nullptr, out, NTOK, HIDDEN, HIDDEN);
}
